// UnrolledMeanShift_29824252903819
// MI455X (gfx1250) — matrix operand order repaired and checked by reading, not compiled, not run on hardware
//
#include <hip/hip_runtime.h>
#include <hip/hip_bf16.h>

typedef __attribute__((ext_vector_type(16))) _Float16 v16h;
typedef __attribute__((ext_vector_type(8)))  _Float16 v8h;
typedef __attribute__((ext_vector_type(8)))  float    v8f;
typedef __attribute__((ext_vector_type(4)))  float    v4f;
typedef __attribute__((ext_vector_type(2)))  int      v2i;

#define IMG_H 256
#define IMG_W 256
#define NDIM  32
#define NITER 3

#if defined(__has_builtin)
#if __has_builtin(__builtin_amdgcn_global_load_async_to_lds_b64)
#define HAVE_ASYNC_LDS 1
#endif
#endif

#ifdef HAVE_ASYNC_LDS
typedef __attribute__((address_space(1))) v2i* global_v2i_ptr;
typedef __attribute__((address_space(3))) v2i* lds_v2i_ptr;
#endif

__global__ __launch_bounds__(32)
void UnrolledMeanShift_29824252903819_kernel(const float* __restrict__ emb,
                                             const float* __restrict__ lbw,
                                             float* __restrict__ out) {
#ifdef HAVE_ASYNC_LDS
    __shared__ __align__(16) float Pstage[NDIM][8][8];  // raw f32 staging (d-major)
#endif
    __shared__ _Float16 Pf16[64][NDIM];   // union patch (u-major), f16
    __shared__ _Float16 PT[NDIM][64];     // transposed copy (d-major) for B frags
    __shared__ float    pp[64];           // ||p_u||^2
    __shared__ _Float16 Zf16[16][NDIM];   // current z, f16
    __shared__ float    Zf32[16][NDIM];   // current z, f32
    __shared__ float    zz[16];           // ||z_n||^2
    __shared__ _Float16 Wld[16][64];      // weights (pixel-major), f16
    __shared__ float    dsum[2][16];
    __shared__ float    denom[16];

    const int lane = threadIdx.x;
    const int lh = lane >> 4;            // lane half (0/1)
    const int ll = lane & 15;
    const int b  = blockIdx.y;
    const int tx = blockIdx.x & 63;
    const int ty = blockIdx.x >> 6;
    const int y0 = ty * 4 - 2;
    const int x0 = tx * 4 - 2;

    const float bw   = log1pf(__expf(lbw[0]));    // softplus
    const float cinv = 1.0f / (2.0f * bw * bw);

    const float* embB = emb + (size_t)b * NDIM * IMG_H * IMG_W;

#ifdef HAVE_ASYNC_LDS
    // ---- Async DMA of the 8x8x32 union patch into LDS (b64 chunks) ----
    // Chunks start at even gx and the image edges are even, so a chunk is
    // either fully inside or fully outside; outside chunks get a safe clamped
    // address and are zeroed in the conversion pass below.
    for (int c = lane; c < NDIM * 8 * 4; c += 32) {
        const int d = c >> 5, rem = c & 31;
        const int uy = rem >> 2, xq = rem & 3;
        const int gy = y0 + uy, gx0 = x0 + 2 * xq;
        const bool ok = ((unsigned)gy < (unsigned)IMG_H) && ((unsigned)gx0 < (unsigned)IMG_W);
        const float* g = ok ? &embB[((size_t)d * IMG_H + gy) * IMG_W + gx0] : embB;
        __builtin_amdgcn_global_load_async_to_lds_b64(
            (global_v2i_ptr)g, (lds_v2i_ptr)&Pstage[d][uy][2 * xq], 0, 0);
    }
#if __has_builtin(__builtin_amdgcn_s_wait_asynccnt)
    __builtin_amdgcn_s_wait_asynccnt(0);
#else
    asm volatile("s_wait_asynccnt 0x0" ::: "memory");
#endif
    __syncthreads();
    // Convert to f16 (+ transposed copy), zeroing out-of-image elements.
    for (int idx = lane; idx < 64 * NDIM; idx += 32) {
        const int u = idx & 63, d = idx >> 6;
        const int uy = u >> 3, ux = u & 7;
        const int gy = y0 + uy, gx = x0 + ux;
        const bool in = ((unsigned)gy < (unsigned)IMG_H) && ((unsigned)gx < (unsigned)IMG_W);
        const float v = in ? Pstage[d][uy][ux] : 0.0f;
        const _Float16 h = (_Float16)v;
        Pf16[u][d] = h;
        PT[d][u]   = h;
    }
#else
    // ---- Fallback: synchronous loads (zero padded), lanes walk x ----
    for (int idx = lane; idx < 64 * NDIM; idx += 32) {
        const int u = idx & 63, d = idx >> 6;
        const int gy = y0 + (u >> 3), gx = x0 + (u & 7);
        float v = 0.0f;
        if ((unsigned)gy < (unsigned)IMG_H && (unsigned)gx < (unsigned)IMG_W)
            v = embB[((size_t)d * IMG_H + gy) * IMG_W + gx];
        const _Float16 h = (_Float16)v;
        Pf16[u][d] = h;
        PT[d][u]   = h;
    }
#endif
    __syncthreads();

    for (int u = lane; u < 64; u += 32) {
        float s = 0.0f;
        #pragma unroll
        for (int d = 0; d < NDIM; ++d) { float v = (float)Pf16[u][d]; s += v * v; }
        pp[u] = s;
    }
    for (int idx = lane; idx < 16 * NDIM; idx += 32) {
        const int n = idx >> 5, d = idx & 31;
        const int u = ((n >> 2) + 2) * 8 + ((n & 3) + 2);
        const float v = (float)Pf16[u][d];
        Zf16[n][d] = (_Float16)v;
        Zf32[n][d] = v;
    }
    __syncthreads();
    if (lane < 16) {
        float s = 0.0f;
        #pragma unroll
        for (int d = 0; d < NDIM; ++d) { float v = (float)Zf16[lane][d]; s += v * v; }
        zz[lane] = s;
    }
    __syncthreads();

    // ---- Hoist all P-derived operands into registers (patches are constant) ----
    v16h paT[4];           // A frags for G = P(64x32) * Z^T
    #pragma unroll
    for (int t = 0; t < 4; ++t)
        #pragma unroll
        for (int j = 0; j < 16; ++j)
            paT[t][j] = Pf16[16 * t + ll][((j < 8) ? j : (j + 8)) + 8 * lh];

    v16h pbT[2][2];        // B frags for numerator = W(16x64) * P(64x32)
    #pragma unroll
    for (int ct = 0; ct < 2; ++ct)
        #pragma unroll
        for (int kk = 0; kk < 2; ++kk)
            #pragma unroll
            for (int j = 0; j < 16; ++j)
                pbT[ct][kk][j] = PT[16 * ct + ll][32 * kk + ((j < 8) ? j : (j + 8)) + 8 * lh];

    float ppv[32];         // pp[u] for this lane's 8 G-rows in each of 4 tiles
    #pragma unroll
    for (int t = 0; t < 4; ++t) {
        v4f p0 = *reinterpret_cast<v4f*>(&pp[16 * t + 8 * lh]);
        v4f p1 = *reinterpret_cast<v4f*>(&pp[16 * t + 8 * lh + 4]);
        #pragma unroll
        for (int r = 0; r < 4; ++r) { ppv[8 * t + r] = p0[r]; ppv[8 * t + 4 + r] = p1[r]; }
    }

    const int py = ll >> 2, px = ll & 3;
    const unsigned rowm = 0x1Fu << py;   // valid uy bits for this pixel's window
    const unsigned colm = 0x1Fu << px;   // valid ux bits

    for (int iter = 0; iter < NITER; ++iter) {
        const float zzn = zz[ll];
        v16h zb;                          // B frag: B(k,n) = Zf16[n][k]
        #pragma unroll
        for (int j = 0; j < 16; ++j) zb[j] = Zf16[ll][((j < 8) ? j : (j + 8)) + 8 * lh];

        // ---- G = P * Z^T, then branchless gaussian weights ----
        float colsum = 0.0f;
        #pragma unroll
        for (int t = 0; t < 4; ++t) {
            v8f g = {};
            g = __builtin_amdgcn_wmma_f32_16x16x32_f16(
                    false, paT[t], false, zb, (short)0, g, false, false);
            const unsigned rbit = (rowm >> (2 * t + lh)) & 1u;  // uy = 2t + lh
            v8h wv;
            #pragma unroll
            for (int r = 0; r < 8; ++r) {
                const float e = __expf((2.0f * g[r] - ppv[8 * t + r] - zzn) * cinv);
                const float w = (rbit & ((colm >> r) & 1u)) ? e : 0.0f;  // ux = r
                colsum += w;
                wv[r] = (_Float16)w;
            }
            *reinterpret_cast<v8h*>(&Wld[ll][16 * t + 8 * lh]) = wv;   // one b128
        }
        dsum[lh][ll] = colsum;
        __syncthreads();
        if (lane < 16) denom[lane] = dsum[0][lane] + dsum[1][lane] + 1e-6f;
        __syncthreads();

        float dnv[8];
        {
            v4f d0 = *reinterpret_cast<v4f*>(&denom[8 * lh]);
            v4f d1 = *reinterpret_cast<v4f*>(&denom[8 * lh + 4]);
            #pragma unroll
            for (int r = 0; r < 4; ++r) { dnv[r] = d0[r]; dnv[4 + r] = d1[r]; }
        }

        // ---- numerator = W * P ; z update ----
        #pragma unroll
        for (int ct = 0; ct < 2; ++ct) {
            v8f acc = {};
            #pragma unroll
            for (int kk = 0; kk < 2; ++kk) {
                v16h wa;
                #pragma unroll
                for (int j = 0; j < 16; ++j)
                    wa[j] = Wld[ll][32 * kk + ((j < 8) ? j : (j + 8)) + 8 * lh];
                acc = __builtin_amdgcn_wmma_f32_16x16x32_f16(
                          false, wa, false, pbT[ct][kk], (short)0, acc, false, false);
            }
            #pragma unroll
            for (int r = 0; r < 8; ++r) {
                const int m = r + 8 * lh;          // pixel
                const int d = 16 * ct + ll;        // dim
                const float znew = acc[r] / dnv[r];
                Zf32[m][d] = znew;
                Zf16[m][d] = (_Float16)znew;
            }
        }
        __syncthreads();
        if (lane < 16) {
            float s = 0.0f;
            #pragma unroll
            for (int d = 0; d < NDIM; ++d) { float v = (float)Zf16[lane][d]; s += v * v; }
            zz[lane] = s;
        }
        __syncthreads();
    }

    // ---- write result (B, D, H, W) ----
    for (int idx = lane; idx < 16 * NDIM; idx += 32) {
        const int d = idx >> 4, n = idx & 15;
        const int gy = ty * 4 + (n >> 2);
        const int gx = tx * 4 + (n & 3);
        out[(((size_t)b * NDIM + d) * IMG_H + gy) * IMG_W + gx] = Zf32[n][d];
    }
}

extern "C" void kernel_launch(void* const* d_in, const int* in_sizes, int n_in,
                              void* d_out, int out_size, void* d_ws, size_t ws_size,
                              hipStream_t stream) {
    const float* emb = (const float*)d_in[0];   // (2, 32, 256, 256) fp32
    const float* lbw = (const float*)d_in[1];   // scalar
    float* out = (float*)d_out;
    dim3 grid(64 * 64, 2);
    dim3 block(32);
    UnrolledMeanShift_29824252903819_kernel<<<grid, block, 0, stream>>>(emb, lbw, out);
}
